// SingleStageDetector_10239202033948
// MI455X (gfx1250) — compile-verified, matrix operand order checked by reading, not run on hardware
//
#include <hip/hip_runtime.h>

typedef __bf16 bf16_t;
typedef bf16_t v16bf __attribute__((ext_vector_type(16)));
typedef float  v8f   __attribute__((ext_vector_type(8)));
typedef unsigned int u32x4 __attribute__((ext_vector_type(4)));
typedef int          i32x4 __attribute__((ext_vector_type(4)));
typedef int          i32x8 __attribute__((ext_vector_type(8)));

#define A_NUM 9
#define C_NUM 20
#define NBOX  40
#define CIN   1280
#define HID   256
#define OUT2  65
#define HW    3136          // 56*56
#define NPIX  64            // pixels per workgroup tile
#define TILES_PER_B 49      // 3136 / 64
#define KB1   40            // CIN/32
#define BLK_FLOATS (32 * NPIX)

// flat offsets of the 5 outputs inside d_out (reference return order)
#define OFF_CONF 0
#define OFF_OFFS 451584
#define OFF_CLS  2257920
#define OFF_PROP 3261440
#define OFF_IOU  5067776

// element j (0..15) of a 16-bit WMMA A/B fragment -> K index, per CDNA5 ISA
// 16-bit 16x32 layout (lanes 0-15: K 0-7,16-23 ; lanes 16-31: +8)
__device__ __forceinline__ int kmap(int j, int g) {
    int v = j >> 1, pos = j & 1;
    int k = (v < 4) ? (v * 2 + pos) : (16 + (v - 4) * 2 + pos);
    return k + g * 8;
}

__device__ __forceinline__ float sigmoidf_(float x) {
    return 1.0f / (1.0f + __expf(-x));
}

// Tensor Data Mover: async 2D block load (32 rows x 64 cols, fp32) into LDS.
// D# layout per CDNA5 ISA 8.3/8.4 (group0: count/lds/global/type, group1:
// data_size=4B, tensor_dim0=HW row, tensor_dim1=CIN, tile 64x32, stride HW).
__device__ __forceinline__ void tdm_load_block(unsigned int lds_addr,
                                               const float* gptr) {
    unsigned long long ga = (unsigned long long)(size_t)gptr;
    u32x4 g0;
    g0[0] = 1u;                                    // count=1 (valid user D#)
    g0[1] = lds_addr;                              // lds_addr [63:32]
    g0[2] = (unsigned int)ga;                      // global_addr lo
    g0[3] = (unsigned int)((ga >> 32) & 0x1FFFFFFu) | (2u << 30); // hi | type=2
    i32x8 g1;
    g1[0] = (int)(2u << 16);                       // data_size = 2 -> 4 bytes
    g1[1] = (int)((HW & 0xFFFFu) << 16);           // tensor_dim0 lo16
    g1[2] = (int)((HW >> 16) | ((CIN & 0xFFFFu) << 16)); // dim0 hi | dim1 lo
    g1[3] = (int)((CIN >> 16) | (NPIX << 16));     // dim1 hi | tile_dim0=64
    g1[4] = 32;                                    // tile_dim1=32, tile_dim2=0
    g1[5] = HW;                                    // tensor_dim0_stride lo32
    g1[6] = 0;                                     // stride hi | dim1_stride lo
    g1[7] = 0;
    i32x4 z4 = {0, 0, 0, 0};
#if defined(__clang_major__) && __clang_major__ >= 23
    i32x8 z8 = {0, 0, 0, 0, 0, 0, 0, 0};
    __builtin_amdgcn_tensor_load_to_lds(g0, g1, z4, z4, z8, 0);
#else
    __builtin_amdgcn_tensor_load_to_lds(g0, g1, z4, z4, 0);
#endif
}

__global__ __launch_bounds__(512)
void det_head_fused(const float* __restrict__ features,
                    const float* __restrict__ grid,
                    const float* __restrict__ anc,
                    const float* __restrict__ bboxes,
                    const float* __restrict__ w1,
                    const float* __restrict__ b1,
                    const float* __restrict__ w2,
                    const float* __restrict__ b2,
                    float* __restrict__ out)
{
    __shared__ float  lds_fbf[2][32][NPIX];        // double-buffered feature block
    __shared__ bf16_t lds_h[NPIX][HID];            // hidden activations [pixel][chan]
    __shared__ float  lds_preds[80][NPIX];         // layer-2 outputs (65 used)
    __shared__ float  lds_prop[A_NUM * NPIX][4];
    __shared__ float  lds_bb[NBOX][4];
    __shared__ float  lds_grid[NPIX][2];

    const int tid  = threadIdx.x;
    const int wave = tid >> 5;
    const int lane = tid & 31;
    const int g    = lane >> 4;
    const int hl   = lane & 15;

    const int blk = blockIdx.x;
    const int b   = blk / TILES_PER_B;
    const int p0  = (blk % TILES_PER_B) * NPIX;

    // ---- stage grid + bboxes for phase 3 (covered by barriers below) ----
    if (tid < NPIX * 2) {
        int j = tid >> 1, c = tid & 1;
        lds_grid[j][c] = grid[((size_t)b * HW + p0 + j) * 2 + c];
    } else if (tid < NPIX * 2 + NBOX * 4) {
        int t = tid - NPIX * 2;
        int n = t >> 2, i = t & 3;
        lds_bb[n][i] = bboxes[((size_t)b * NBOX + n) * 5 + i];
    }

    const float* fbase = features + (size_t)b * CIN * HW + p0;
    const unsigned int lds_fb_base = (unsigned int)(size_t)&lds_fbf[0][0][0];

    // TDM prologue: kick off K-block 0
    if (wave == 0) tdm_load_block(lds_fb_base, fbase);

    // =========================== Phase 1: GEMM1 ===========================
    const int m1 = wave * 16 + hl;                 // hidden row this lane feeds
    v8f acc[4] = {};

    for (int kb = 0; kb < KB1; ++kb) {
        const int cur = kb & 1;
        __syncthreads();                           // prev buffer fully consumed
        if (wave == 0) {
            if (kb + 1 < KB1) {
                tdm_load_block(lds_fb_base + (unsigned)(1 - cur) * BLK_FLOATS * 4,
                               fbase + (size_t)(kb + 1) * 32 * HW);
                __builtin_amdgcn_s_wait_tensorcnt(1);  // current block landed
            } else {
                __builtin_amdgcn_s_wait_tensorcnt(0);
            }
        }
        __syncthreads();

        // prefetch next w1 K-slab while computing this one
        if (kb + 1 < KB1)
            __builtin_prefetch(&w1[(size_t)m1 * CIN + (kb + 1) * 32], 0, 0);

        v16bf af;
#pragma unroll
        for (int j = 0; j < 16; ++j)
            af[j] = (bf16_t)w1[(size_t)m1 * CIN + kb * 32 + kmap(j, g)];

#pragma unroll
        for (int nt = 0; nt < 4; ++nt) {
            v16bf bfv;
#pragma unroll
            for (int j = 0; j < 16; ++j)
                bfv[j] = (bf16_t)lds_fbf[cur][kmap(j, g)][nt * 16 + hl];
            acc[nt] = __builtin_amdgcn_wmma_f32_16x16x32_bf16(
                          false, af, false, bfv, (short)0, acc[nt], false, false);
        }
    }
    __syncthreads();

    // bias + LeakyReLU, store hidden tile to LDS as bf16 ([pixel][chan])
#pragma unroll
    for (int nt = 0; nt < 4; ++nt) {
        int px = nt * 16 + hl;
#pragma unroll
        for (int r = 0; r < 8; ++r) {
            int m = wave * 16 + r + g * 8;
            float v = acc[nt][r] + b1[m];
            v = v > 0.0f ? v : 0.01f * v;
            lds_h[px][m] = (bf16_t)v;
        }
    }
    __syncthreads();

    // =========================== Phase 2: GEMM2 ===========================
    // 20 wave-tiles: 5 M-tiles (80 rows >= 65) x 4 N-tiles, over 16 waves
    for (int t = wave; t < 20; t += 16) {
        const int mt = t % 5, nt = t / 5;
        const int m2 = mt * 16 + hl;
        const int px = nt * 16 + hl;
        v8f acc2 = {};
        for (int kb = 0; kb < HID / 32; ++kb) {
            v16bf af, bfv;
#pragma unroll
            for (int j = 0; j < 16; ++j) {
                int k = kb * 32 + kmap(j, g);
                af[j]  = (m2 < OUT2) ? (bf16_t)w2[(size_t)m2 * HID + k]
                                     : (bf16_t)0.0f;
                bfv[j] = lds_h[px][k];
            }
            acc2 = __builtin_amdgcn_wmma_f32_16x16x32_bf16(
                       false, af, false, bfv, (short)0, acc2, false, false);
        }
#pragma unroll
        for (int r = 0; r < 8; ++r) {
            int m = mt * 16 + r + g * 8;
            float bias = (m < OUT2) ? b2[m] : 0.0f;
            lds_preds[m][nt * 16 + hl] = acc2[r] + bias;
        }
    }
    __syncthreads();

    // ======================= Phase 3: post-processing ======================
    for (int it = tid; it < A_NUM * NPIX; it += 512) {  // (anchor, pixel) items
        int a = it >> 6, j = it & 63;
        int pix = p0 + j;
        float t0 = lds_preds[a * 5 + 0][j];
        float t1 = lds_preds[a * 5 + 1][j];
        float t2 = lds_preds[a * 5 + 2][j];
        float t3 = lds_preds[a * 5 + 3][j];
        float t4 = lds_preds[a * 5 + 4][j];

        float conf = sigmoidf_(t0);
        float ox = sigmoidf_(t1) - 0.5f;
        float oy = sigmoidf_(t2) - 0.5f;

        out[OFF_CONF + ((size_t)b * A_NUM + a) * HW + pix] = conf;

        size_t ob = OFF_OFFS + (((size_t)b * A_NUM + a) * 4) * (size_t)HW + pix;
        out[ob]          = ox;
        out[ob + HW]     = oy;
        out[ob + 2 * HW] = t3;
        out[ob + 3 * HW] = t4;

        float gx = lds_grid[j][0], gy = lds_grid[j][1];
        float aw = fabsf(anc[a * 2]), ah = fabsf(anc[a * 2 + 1]);
        float cx = gx + ox, cy = gy + oy;
        float pw = aw * __expf(t3), ph = ah * __expf(t4);
        float x0 = cx - 0.5f * pw, y0 = cy - 0.5f * ph;
        float x1 = cx + 0.5f * pw, y1 = cy + 0.5f * ph;

        size_t pb = OFF_PROP + (((size_t)b * A_NUM + a) * HW + pix) * 4;
        out[pb + 0] = x0; out[pb + 1] = y0; out[pb + 2] = x1; out[pb + 3] = y1;
        lds_prop[it][0] = x0; lds_prop[it][1] = y0;
        lds_prop[it][2] = x1; lds_prop[it][3] = y1;
    }
    for (int e = tid; e < C_NUM * NPIX; e += 512) {     // class scores copy
        int c = e >> 6, j = e & 63;
        out[OFF_CLS + ((size_t)b * C_NUM + c) * HW + p0 + j] =
            lds_preds[5 * A_NUM + c][j];
    }
    __syncthreads();

    // IoU matrix: 576 proposals x 40 boxes
    for (int e = tid; e < A_NUM * NPIX * NBOX; e += 512) {
        int n = e % NBOX;
        int item = e / NBOX;
        int a = item >> 6, j = item & 63;
        float px0 = lds_prop[item][0], py0 = lds_prop[item][1];
        float px1 = lds_prop[item][2], py1 = lds_prop[item][3];
        float bx0 = lds_bb[n][0], by0 = lds_bb[n][1];
        float bx1 = lds_bb[n][2], by1 = lds_bb[n][3];
        float ix = fmaxf(fmaxf(bx1 - px0, 0.f) - fmaxf(bx1 - px1, 0.f) - fmaxf(bx0 - px0, 0.f), 0.f);
        float iy = fmaxf(fmaxf(by1 - py0, 0.f) - fmaxf(by1 - py1, 0.f) - fmaxf(by0 - py0, 0.f), 0.f);
        float inter = ix * iy;
        float uni = (px1 - px0) * (py1 - py0) + (bx1 - bx0) * (by1 - by0) - inter;
        out[OFF_IOU + ((size_t)b * (A_NUM * HW) + (size_t)a * HW + (p0 + j)) * NBOX + n] =
            inter / uni;
    }
}

extern "C" void kernel_launch(void* const* d_in, const int* in_sizes, int n_in,
                              void* d_out, int out_size, void* d_ws, size_t ws_size,
                              hipStream_t stream) {
    const float* features = (const float*)d_in[0];
    const float* grid     = (const float*)d_in[1];
    const float* anc      = (const float*)d_in[2];
    const float* bboxes   = (const float*)d_in[3];
    const float* w1       = (const float*)d_in[4];
    const float* b1       = (const float*)d_in[5];
    const float* w2       = (const float*)d_in[6];
    const float* b2       = (const float*)d_in[7];
    float* out = (float*)d_out;

    dim3 gridDim(16 * TILES_PER_B);   // 784 workgroups, one 64-pixel tile each
    dim3 blockDim(512);               // 16 waves (wave32)
    det_head_fused<<<gridDim, blockDim, 0, stream>>>(
        features, grid, anc, bboxes, w1, b1, w2, b2, out);
}
